// NonEquivariantDecoderUpdate_25881472925818
// MI455X (gfx1250) — matrix operand order repaired and checked by reading, not compiled, not run on hardware
//
#include <hip/hip_runtime.h>
#include <hip/hip_bf16.h>
#include <math.h>

typedef __attribute__((ext_vector_type(16))) __bf16 v16bf;
typedef __attribute__((ext_vector_type(8)))  float  v8f;

#define NROW   32768
#define DIM    256
#define DFF    512
#define NBATCH 64
#define NCHAIN 256

// ---- workspace byte offsets (all 256B aligned) ----
#define OFF_WTG   0u                        // 4 gate weights, bf16 [512][256] each (transposed)
#define SZ_WTG    (4u*DFF*DIM*2u)           // 1,048,576
#define OFF_WTO   (OFF_WTG + SZ_WTG)        // W_out transposed bf16 [256][512]
#define SZ_WTO    (DIM*DFF*2u)              // 262,144
#define OFF_SB    (OFF_WTO + SZ_WTO)        // batch segment sums f32 [64][512]
#define SZ_SB     (NBATCH*DFF*4u)
#define OFF_SC    (OFF_SB + SZ_SB)          // chain segment sums f32 [256][512]
#define SZ_SC     (NCHAIN*DFF*4u)
#define OFF_CNTB  (OFF_SC + SZ_SC)          // f32 [64]
#define OFF_CNTC  (OFF_CNTB + 256u)         // f32 [256]
#define OFF_MB    (OFF_CNTC + 1024u)        // means bf16 [64][512]
#define OFF_MC    (OFF_MB + NBATCH*DFF*2u)  // means bf16 [256][512]
#define OFF_PL    (OFF_MC + NCHAIN*DFF*2u)  // local part bf16 [N][512]

__device__ __forceinline__ float gelu_tanh(float x) {
    float x3 = x * x * x;
    return 0.5f * x * (1.0f + tanhf(0.7978845608028654f * (x + 0.044715f * x3)));
}

__device__ __forceinline__ void atomAddF(float* p, float v) {
    unsafeAtomicAdd(p, v);   // lowers to global_atomic_add_f32
}

// ---------------- kernel 0: zero Sb/Sc/counts (contiguous region) ---------------
__global__ void k_zero(float* p, int n) {
    int i = blockIdx.x * blockDim.x + threadIdx.x;
    if (i < n) p[i] = 0.0f;
}

// ---------------- kernel 1: transpose + bf16-convert all weights ----------------
__global__ void k_prep(const float* __restrict__ Wu, const float* __restrict__ Wl,
                       const float* __restrict__ Wc, const float* __restrict__ Wb,
                       const float* __restrict__ Wo,
                       __bf16* __restrict__ WtG, __bf16* __restrict__ WtO) {
    int i = blockIdx.x * blockDim.x + threadIdx.x;
    const int G = 4 * DFF * DIM;
    if (i < G) {
        int m = i / (DFF * DIM);
        int r = i % (DFF * DIM);
        int n = r / DIM, k = r % DIM;
        const float* W = (m == 0) ? Wu : (m == 1) ? Wl : (m == 2) ? Wc : Wb;
        WtG[i] = (__bf16)W[k * DFF + n];
    } else if (i < G + DIM * DFF) {
        int j = i - G;
        int n = j / DFF, k = j % DFF;
        WtO[j] = (__bf16)Wo[k * DIM + n];
    }
}

// ---------------- kernel 2: segment counts ----------------
__global__ void k_counts(const unsigned char* __restrict__ mask,
                         const long long* __restrict__ batch,
                         const long long* __restrict__ chain,
                         float* __restrict__ cntB, float* __restrict__ cntC) {
    int i = blockIdx.x * blockDim.x + threadIdx.x;
    if (i < NROW) {
        float m = mask[i] ? 1.0f : 0.0f;
        atomAddF(&cntB[(int)batch[i]], m);
        atomAddF(&cntC[(int)chain[i]], m);
    }
}

// ---------------- kernel 3: fused 4-GEMM + gelu + segment accumulate ------------
// block = 256 rows x 64 cols (one of 8 column groups), 512 threads = 16 waves
// arranged 4 wave-rows x 4 col-tiles. A tile (128 KB) and the full B slab for
// these 64 cols x 4 matrices (128 KB) live in LDS -> each B fragment is loaded
// from global exactly once per block and reused 16x (4 row-tiles x 4 wave-rows).
__global__ void __launch_bounds__(512)
k_main(const float* __restrict__ local,
       const unsigned char* __restrict__ mask,
       const long long* __restrict__ batch,
       const long long* __restrict__ chain,
       const __bf16* __restrict__ WtG,   // [4][512][256] bf16
       float* __restrict__ Sb, float* __restrict__ Sc,
       __bf16* __restrict__ Pl) {
    __shared__ __bf16 As[256][DIM];      // 128 KB, A tile bf16
    __shared__ __bf16 Bs[4][64][DIM];    // 128 KB, B slab bf16
    __shared__ long long sB[256], sC[256];
    __shared__ float sM[256];

    const int tid    = threadIdx.x;
    const int rowblk = blockIdx.x >> 3;      // 128 row groups
    const int colgrp = blockIdx.x & 7;       // 8 column groups
    const int row0   = rowblk * 256;
    const int col0   = colgrp * 64;

    // stage A tile fp32 -> bf16 (256 x 256), vectorized
    {
        const float4* Ls = (const float4*)(local + (size_t)row0 * DIM);
        __bf16* Ad = &As[0][0];
        for (int q = tid; q < 256 * DIM / 4; q += 512) {
            float4 f = Ls[q];
            union { __bf16 h[4]; uint2 u; } p;
            p.h[0] = (__bf16)f.x; p.h[1] = (__bf16)f.y;
            p.h[2] = (__bf16)f.z; p.h[3] = (__bf16)f.w;
            *(uint2*)(Ad + q * 4) = p.u;
        }
    }
    // stage B slab (4 matrices x 64 cols x 256 K bf16), straight b128 copies
    for (int q = tid; q < 8192; q += 512) {
        int m = q >> 11, rem = q & 2047;
        int col = rem >> 5, part = rem & 31;
        const uint4* src = (const uint4*)(WtG + (size_t)m * DFF * DIM
                                          + (size_t)(col0 + col) * DIM + part * 8);
        *(uint4*)(&Bs[m][col][part * 8]) = *src;
    }
    if (tid < 256) {
        sB[tid] = batch[row0 + tid];
        sC[tid] = chain[row0 + tid];
        sM[tid] = mask[row0 + tid] ? 1.0f : 0.0f;
    }
    __syncthreads();

    const int w    = tid >> 5;           // wave id 0..15
    const int lane = tid & 31;
    const int nloc = lane & 15;
    const int half = lane >> 4;
    const int wr   = w >> 2;             // wave-row 0..3  (64 rows each)
    const int ct   = w & 3;              // col-tile 0..3  (16 cols each)
    const int rbase = wr * 64;

    v8f acc[4][4];                       // [matrix][row-tile]
    #pragma unroll
    for (int m = 0; m < 4; ++m)
        #pragma unroll
        for (int rt = 0; rt < 4; ++rt) acc[m][rt] = (v8f)(0.0f);

    for (int kc = 0; kc < DIM / 32; ++kc) {
        union { unsigned int u[8]; v16bf v; } au[4];
        #pragma unroll
        for (int rt = 0; rt < 4; ++rt) {
            #pragma unroll
            for (int v = 0; v < 8; ++v) {
                int koff = v * 2 + half * 16;
                au[rt].u[v] = *(const unsigned int*)&As[rbase + rt * 16 + nloc][kc * 32 + koff];
            }
        }
        #pragma unroll
        for (int m = 0; m < 4; ++m) {
            const __bf16* bp = &Bs[m][ct * 16 + nloc][kc * 32 + half * 16];
            union { uint4 q[2]; v16bf v; } bu;
            bu.q[0] = *(const uint4*)bp;
            bu.q[1] = *(const uint4*)(bp + 8);
            #pragma unroll
            for (int rt = 0; rt < 4; ++rt) {
                acc[m][rt] = __builtin_amdgcn_wmma_f32_16x16x32_bf16(
                    false, au[rt].v, false, bu.v, (short)0, acc[m][rt], false, false);
            }
        }
    }

    // epilogue: gelu gates, local-part store, run-length segment atomics.
    const int col = col0 + ct * 16 + nloc;
    long long curB = sB[rbase + half * 8], curC = sC[rbase + half * 8];
    float aB = 0.0f, aC = 0.0f;
    #pragma unroll
    for (int rt = 0; rt < 4; ++rt) {
        #pragma unroll
        for (int r = 0; r < 8; ++r) {
            int row = rbase + rt * 16 + half * 8 + r;
            float upd = acc[0][rt][r];
            float msk = sM[row];
            float pl = gelu_tanh(acc[1][rt][r]) * upd;
            float pc = gelu_tanh(acc[2][rt][r]) * upd * msk;
            float pb = gelu_tanh(acc[3][rt][r]) * upd * msk;
            Pl[(size_t)(row0 + row) * DFF + col] = (__bf16)pl;
            long long ib = sB[row];
            if (ib != curB) { atomAddF(Sb + (size_t)curB * DFF + col, aB); aB = 0.0f; curB = ib; }
            aB += pb;
            long long ic = sC[row];
            if (ic != curC) { atomAddF(Sc + (size_t)curC * DFF + col, aC); aC = 0.0f; curC = ic; }
            aC += pc;
        }
    }
    atomAddF(Sb + (size_t)curB * DFF + col, aB);
    atomAddF(Sc + (size_t)curC * DFF + col, aC);
}

// ---------------- kernel 4: segment means -> bf16 ----------------
__global__ void k_means(const float* __restrict__ Sb, const float* __restrict__ Sc,
                        const float* __restrict__ cntB, const float* __restrict__ cntC,
                        __bf16* __restrict__ Mb, __bf16* __restrict__ Mc) {
    int i = blockIdx.x * blockDim.x + threadIdx.x;
    if (i < NBATCH * DFF) {
        Mb[i] = (__bf16)(Sb[i] / fmaxf(cntB[i / DFF], 1e-6f));
    } else if (i < (NBATCH + NCHAIN) * DFF) {
        int j = i - NBATCH * DFF;
        Mc[j] = (__bf16)(Sc[j] / fmaxf(cntC[j / DFF], 1e-6f));
    }
}

// ---------------- kernel 5: final GEMM hidden @ W_out + b ----------------
// block = 64 rows x 256 out cols, 512 threads = 16 waves.
// wave w -> one 16-col tile x 4 row-tiles; B fragment reused 4x.
__global__ void __launch_bounds__(512)
k_final(const __bf16* __restrict__ Pl,
        const __bf16* __restrict__ Mb, const __bf16* __restrict__ Mc,
        const long long* __restrict__ batch, const long long* __restrict__ chain,
        const __bf16* __restrict__ WtO,      // [256][512] bf16
        const float* __restrict__ bOut,
        float* __restrict__ out) {
    __shared__ __bf16 Hs[64][DFF];       // 64 KB hidden tile bf16
    __shared__ int sBi[64], sCi[64];

    const int tid  = threadIdx.x;
    const int row0 = blockIdx.x * 64;

    if (tid < 64) {
        sBi[tid] = (int)batch[row0 + tid];
        sCi[tid] = (int)chain[row0 + tid];
    }
    __syncthreads();

    #pragma unroll
    for (int e = tid; e < 64 * DFF; e += 512) {
        int r = e >> 9, c = e & 511;
        float h = (float)Pl[(size_t)(row0 + r) * DFF + c]
                + (float)Mb[(size_t)sBi[r] * DFF + c]
                + (float)Mc[(size_t)sCi[r] * DFF + c];
        Hs[r][c] = (__bf16)h;
    }
    __syncthreads();

    const int w    = tid >> 5;
    const int lane = tid & 31;
    const int nloc = lane & 15;
    const int half = lane >> 4;
    const int colb = w * 16;

    v8f acc[4];
    #pragma unroll
    for (int rt = 0; rt < 4; ++rt) acc[rt] = (v8f)(0.0f);

    for (int kc = 0; kc < DFF / 32; ++kc) {
        union { unsigned int u[8]; v16bf v; } au[4];
        #pragma unroll
        for (int rt = 0; rt < 4; ++rt) {
            #pragma unroll
            for (int v = 0; v < 8; ++v) {
                int koff = v * 2 + half * 16;
                au[rt].u[v] = *(const unsigned int*)&Hs[rt * 16 + nloc][kc * 32 + koff];
            }
        }
        const __bf16* bp = WtO + (size_t)(colb + nloc) * DFF + kc * 32 + half * 16;
        union { uint4 q[2]; v16bf v; } bu;
        bu.q[0] = *(const uint4*)bp;
        bu.q[1] = *(const uint4*)(bp + 8);
        #pragma unroll
        for (int rt = 0; rt < 4; ++rt) {
            acc[rt] = __builtin_amdgcn_wmma_f32_16x16x32_bf16(
                false, au[rt].v, false, bu.v, (short)0, acc[rt], false, false);
        }
    }

    const int col  = colb + nloc;
    const float bias = bOut[col];
    #pragma unroll
    for (int rt = 0; rt < 4; ++rt) {
        #pragma unroll
        for (int r = 0; r < 8; ++r) {
            int row = rt * 16 + half * 8 + r;
            out[(size_t)(row0 + row) * DIM + col] = acc[rt][r] + bias;
        }
    }
}

extern "C" void kernel_launch(void* const* d_in, const int* in_sizes, int n_in,
                              void* d_out, int out_size, void* d_ws, size_t ws_size,
                              hipStream_t stream) {
    const float*         local = (const float*)d_in[0];
    const unsigned char* mask  = (const unsigned char*)d_in[1];
    const long long*     batch = (const long long*)d_in[2];
    const long long*     chain = (const long long*)d_in[3];
    const float* Wu = (const float*)d_in[4];
    const float* Wl = (const float*)d_in[5];
    const float* Wc = (const float*)d_in[6];
    const float* Wb = (const float*)d_in[7];
    const float* Wo = (const float*)d_in[8];
    const float* bO = (const float*)d_in[9];
    float* out = (float*)d_out;

    char* ws = (char*)d_ws;
    __bf16* WtG  = (__bf16*)(ws + OFF_WTG);
    __bf16* WtO  = (__bf16*)(ws + OFF_WTO);
    float*  Sb   = (float*)(ws + OFF_SB);
    float*  Sc   = (float*)(ws + OFF_SC);
    float*  cntB = (float*)(ws + OFF_CNTB);
    float*  cntC = (float*)(ws + OFF_CNTC);
    __bf16* Mb   = (__bf16*)(ws + OFF_MB);
    __bf16* Mc   = (__bf16*)(ws + OFF_MC);
    __bf16* Pl   = (__bf16*)(ws + OFF_PL);

    // 0) zero contiguous [Sb | Sc | cnts] region
    int nz = (int)((OFF_MB - OFF_SB) / 4);
    k_zero<<<(nz + 255) / 256, 256, 0, stream>>>((float*)(ws + OFF_SB), nz);

    // 1) weight transpose + bf16
    int np = 4 * DFF * DIM + DIM * DFF;
    k_prep<<<(np + 255) / 256, 256, 0, stream>>>(Wu, Wl, Wc, Wb, Wo, WtG, WtO);

    // 2) segment counts
    k_counts<<<NROW / 256, 256, 0, stream>>>(mask, batch, chain, cntB, cntC);

    // 3) fused gate GEMMs + segment accumulation (256-row x 64-col blocks)
    k_main<<<(NROW / 256) * 8, 512, 0, stream>>>(local, mask, batch, chain, WtG, Sb, Sc, Pl);

    // 4) means
    int nm = (NBATCH + NCHAIN) * DFF;
    k_means<<<(nm + 255) / 256, 256, 0, stream>>>(Sb, Sc, cntB, cntC, Mb, Mc);

    // 5) final GEMM + bias (64-row blocks)
    k_final<<<NROW / 64, 512, 0, stream>>>(Pl, Mb, Mc, batch, chain, WtO, bO, out);
}
